// WindowAttention_21088289423772
// MI455X (gfx1250) — matrix operand order repaired and checked by reading, not compiled, not run on hardware
//
#include <hip/hip_runtime.h>

typedef __attribute__((ext_vector_type(16))) __bf16 v16bf;
typedef __attribute__((ext_vector_type(2)))  __bf16 v2bf;
typedef __attribute__((ext_vector_type(8)))  float  v8f;
typedef unsigned short u16_t;
typedef unsigned int   u32_t;

union FragAB { v16bf v; uint4 q[2]; };

// Packed f32x2 -> bf16x2 (RNE); clang matches this to a single v_cvt_pk_bf16_f32
__device__ __forceinline__ u32_t pk2(float a, float b) {
  v2bf h;
  h[0] = (__bf16)a;
  h[1] = (__bf16)b;
  return __builtin_bit_cast(u32_t, h);
}

// A-fragment (16x32 bf16, row-major source [M][K], row stride rs in elements).
// lane: row = M-tile row + lane%16; VGPR0-3 hold K = k0+8*half .. +7, VGPR4-7 hold +16.
__device__ __forceinline__ v16bf ldA(const u16_t* base, int row, int rs, int k0, int half) {
  FragAB f;
  const u16_t* p = base + row * rs + k0 + half * 8;
  f.q[0] = *(const uint4*)(p);
  f.q[1] = *(const uint4*)(p + 16);
  return f.v;
}
// B-fragment (32x16 bf16) loaded from B^T stored row-major [N][K], row stride rs.
// lane: col = N-tile col + lane%16; VGPR0-3 hold K = k0+8*half .. +7, VGPR4-7 hold +16.
__device__ __forceinline__ v16bf ldB(const u16_t* base, int col, int rs, int k0, int half) {
  FragAB f;
  const u16_t* p = base + col * rs + k0 + half * 8;
  f.q[0] = *(const uint4*)(p);
  f.q[1] = *(const uint4*)(p + 16);
  return f.v;
}
__device__ __forceinline__ v8f wmma_bf(v16bf a, v16bf b, v8f c) {
  return __builtin_amdgcn_wmma_f32_16x16x32_bf16(false, a, false, b, (short)0, c, false, false);
}
// Store D-fragment transposed-contiguous: dest laid out [Ncol][Mrow] (row stride rs),
// lane writes its 8 row-values (M = row0 + 8*half + v) for its fixed column as one b128.
__device__ __forceinline__ void stDT(u16_t* base, int rs, int col, int row0, int half, v8f a) {
  uint4 u;
  u.x = pk2(a[0], a[1]); u.y = pk2(a[2], a[3]);
  u.z = pk2(a[4], a[5]); u.w = pk2(a[6], a[7]);
  *(uint4*)(base + col * rs + row0 + half * 8) = u;
}

// ---------------- weight pre-conversion: f32 -> bf16 into d_ws ----------------
__global__ __launch_bounds__(256) void wconv_kernel(
    const float* __restrict__ wq, const float* __restrict__ wk,
    const float* __restrict__ wv, const float* __restrict__ wo,
    u16_t* __restrict__ dst) {
  // 64 blocks x 256 threads; each thread converts 4 elements of each matrix (float4 in, uint2 out)
  int i = blockIdx.x * blockDim.x + threadIdx.x;    // 0 .. 16383
  const float4 fq = ((const float4*)wq)[i];
  const float4 fk = ((const float4*)wk)[i];
  const float4 fv = ((const float4*)wv)[i];
  const float4 fo = ((const float4*)wo)[i];
  uint2* d = (uint2*)dst;
  d[i]            = make_uint2(pk2(fq.x, fq.y), pk2(fq.z, fq.w));
  d[4096 + i]     = make_uint2(pk2(fk.x, fk.y), pk2(fk.z, fk.w));
  d[8192 + i]     = make_uint2(pk2(fv.x, fv.y), pk2(fv.z, fv.w));
  d[12288 + i]    = make_uint2(pk2(fo.x, fo.y), pk2(fo.z, fo.w));
}

// ---------------- main fused attention kernel: 1 block = 1 batch window ----------------
__global__ __launch_bounds__(256) void WindowAttention_21088289423772_kernel(
    const float* __restrict__ x,   const float* __restrict__ ent,
    const u16_t* __restrict__ Wb,  // bf16: [Wq|Wk|Wv|Wo], each 128x128 row-major [d][c]
    const float* __restrict__ bq,  const float* __restrict__ bk,
    const float* __restrict__ bv,  const float* __restrict__ bo,
    const int*   __restrict__ betap, float* __restrict__ out) {
  __shared__ __align__(16) u16_t smem[49152];   // 96 KB
  __shared__ float sG[64];
  u16_t* sX  = smem;            // x   bf16 [n][c]   rs=128  (reused as Y later)
  u16_t* sQ  = smem + 8192;     // Q        [n][d]   rs=128
  u16_t* sK  = smem + 16384;    // K        [m][d]   rs=128
  u16_t* sVT = smem + 24576;    // V^T      [d][m]   rs=64
  u16_t* sP  = smem + 32768;    // attn     [h][n][m] rs=64 (4 heads x 8KB)
  u16_t* sY  = sX;              // Y = attn@V  [n][c] rs=128 (aliases dead X)

  const int tid  = threadIdx.x;
  const int lane = tid & 31;
  const int wave = tid >> 5;
  const int ln   = lane & 15;
  const int half = lane >> 4;
  const int b    = blockIdx.x;

  // Warm the WGP path with the (L2-resident) bf16 weights: 256 threads x 512B = 128 KB
  __builtin_prefetch((const char*)Wb + tid * 512, 0, 1);   // -> global_prefetch_b8

  // ---- Phase A: stage x (f32 -> bf16) into LDS; gating g[m] = exp(-beta*entropy) ----
  {
    const float4* xs = (const float4*)(x + (size_t)b * 8192);
    u32_t* xd = (u32_t*)sX;
#pragma unroll
    for (int i = 0; i < 8; ++i) {
      float4 f = xs[tid * 8 + i];
      xd[tid * 16 + i * 2 + 0] = pk2(f.x, f.y);
      xd[tid * 16 + i * 2 + 1] = pk2(f.z, f.w);
    }
    if (tid < 64) {
      float beta = (float)(*betap);
      sG[tid] = __expf(-beta * ent[b * 64 + tid]);
    }
  }
  __syncthreads();

  // ---- Phase B: Q^T = Wq@x^T, K^T = Wk@x^T (stored as Q,K row-major); V = x@Wv^T (stored as V^T) ----
  // Branch-free: compile-time mat unroll (Q,K), then V. EXEC stays all-1s around every WMMA.
#pragma unroll
  for (int mat = 0; mat < 2; ++mat) {
    const u16_t* W  = Wb + mat * 16384;
    const float* bi = (mat == 0) ? bq : bk;
    u16_t* D = (mat == 0) ? sQ : sK;
    for (int t = wave; t < 32; t += 8) {
      const int Mt = t & 7, Nt = t >> 3; // M over d (8 tiles), N over tokens (4 tiles)
      v8f acc;
      const float* bp = bi + Mt * 16 + half * 8;
#pragma unroll
      for (int v = 0; v < 8; ++v) acc[v] = bp[v];
#pragma unroll
      for (int kk = 0; kk < 4; ++kk) {
        v16bf a  = ldA(W,  Mt * 16 + ln, 128, kk * 32, half);   // W rows d (global/L2)
        v16bf bb = ldB(sX, Nt * 16 + ln, 128, kk * 32, half);   // B^T = x rows n (LDS)
        acc = wmma_bf(a, bb, acc);
      }
      stDT(D, 128, Nt * 16 + ln, Mt * 16, half, acc);           // -> [token][d]
    }
  }
  for (int t = wave; t < 32; t += 8) {   // V computed forward -> stores as V^T
    const u16_t* W = Wb + 2 * 16384;
    const int Mt = t & 3, Nt = t >> 2;   // M over tokens (4), N over d (8)
    float bvv = bv[Nt * 16 + ln];
    v8f acc;
#pragma unroll
    for (int v = 0; v < 8; ++v) acc[v] = bvv;
#pragma unroll
    for (int kk = 0; kk < 4; ++kk) {
      v16bf a  = ldA(sX, Mt * 16 + ln, 128, kk * 32, half);     // x rows n
      v16bf bb = ldB(W,  Nt * 16 + ln, 128, kk * 32, half);     // B^T = Wv rows d
      acc = wmma_bf(a, bb, acc);
    }
    stDT(sVT, 64, Nt * 16 + ln, Mt * 16, half, acc);            // -> [d][m]
  }
  __syncthreads();

  // ---- Phase C: S^T = K@Q^T per (head, query-strip); fused softmax+gating+renorm; store P [n][m] ----
  for (int job = wave; job < 16; job += 8) {
    const int h  = job >> 2;
    const int Nt = job & 3;
    const int n  = Nt * 16 + ln;
    v16bf qb = ldB(sQ, n, 128, h * 32, half);                   // shared across 4 key tiles
    v8f s[4];
#pragma unroll
    for (int t = 0; t < 4; ++t) {
      v16bf a = ldA(sK, t * 16 + ln, 128, h * 32, half);
      v8f z;
#pragma unroll
      for (int v = 0; v < 8; ++v) z[v] = 0.0f;
      s[t] = wmma_bf(a, qb, z);                                 // K=HD=32: one WMMA/tile
    }
    const float scale = 0.17677669529663687f;                   // 1/sqrt(32)
    float mx = -3.0e38f;
#pragma unroll
    for (int t = 0; t < 4; ++t)
#pragma unroll
      for (int v = 0; v < 8; ++v) { s[t][v] *= scale; mx = fmaxf(mx, s[t][v]); }
    mx = fmaxf(mx, __shfl_xor(mx, 16, 32));                     // lane-half pair holds full column
    float sum = 0.0f;
#pragma unroll
    for (int t = 0; t < 4; ++t) {
      const float* gp = sG + t * 16 + half * 8;
#pragma unroll
      for (int v = 0; v < 8; ++v) {
        float p = __expf(s[t][v] - mx) * gp[v];                 // gate by key entropy
        s[t][v] = p; sum += p;
      }
    }
    sum += __shfl_xor(sum, 16, 32);
    float r = 1.0f / (sum + 1e-9f);
    u16_t* Ph = sP + h * 4096;
#pragma unroll
    for (int t = 0; t < 4; ++t) {
      v8f p;
#pragma unroll
      for (int v = 0; v < 8; ++v) p[v] = s[t][v] * r;
      stDT(Ph, 64, n, t * 16, half, p);                         // -> P [n][m]
    }
  }
  __syncthreads();

  // ---- Phase D: Z^T = V^T @ P^T per head; store as Y [n][c] ----
  for (int job = wave; job < 32; job += 8) {
    const int h  = job >> 3;
    const int rr = job & 7;
    const int Mt = rr & 1, Nt = rr >> 1;   // M over head-dim (2 tiles), N over queries (4)
    const int n = Nt * 16 + ln;
    const u16_t* Ph = sP + h * 4096;
    v8f acc;
#pragma unroll
    for (int v = 0; v < 8; ++v) acc[v] = 0.0f;
#pragma unroll
    for (int kk = 0; kk < 2; ++kk) {
      v16bf a  = ldA(sVT, h * 32 + Mt * 16 + ln, 64, kk * 32, half);
      v16bf bb = ldB(Ph, n, 64, kk * 32, half);                 // B^T = P rows n
      acc = wmma_bf(a, bb, acc);
    }
    stDT(sY, 128, n, h * 32 + Mt * 16, half, acc);              // -> Y [n][c]
  }
  __syncthreads();

  // ---- Phase E: Out^T = Wo @ Y^T + bo; f32 stores straight to global [n][d] ----
  for (int job = wave; job < 32; job += 8) {
    const int Mt = job & 7, Nt = job >> 3;
    const int n = Nt * 16 + ln;
    const u16_t* W = Wb + 3 * 16384;
    v8f acc;
    const float* bp = bo + Mt * 16 + half * 8;
#pragma unroll
    for (int v = 0; v < 8; ++v) acc[v] = bp[v];
#pragma unroll
    for (int kk = 0; kk < 4; ++kk) {
      v16bf a  = ldA(W,  Mt * 16 + ln, 128, kk * 32, half);
      v16bf bb = ldB(sY, n, 128, kk * 32, half);
      acc = wmma_bf(a, bb, acc);
    }
    float* op = out + ((size_t)b * 64 + n) * 128 + Mt * 16 + half * 8;
    *(float4*)(op)     = make_float4(acc[0], acc[1], acc[2], acc[3]);
    *(float4*)(op + 4) = make_float4(acc[4], acc[5], acc[6], acc[7]);
  }
}

extern "C" void kernel_launch(void* const* d_in, const int* in_sizes, int n_in,
                              void* d_out, int out_size, void* d_ws, size_t ws_size,
                              hipStream_t stream) {
  const float* x   = (const float*)d_in[0];
  const float* ent = (const float*)d_in[1];
  const float* Wq  = (const float*)d_in[2];
  const float* bq  = (const float*)d_in[3];
  const float* Wk  = (const float*)d_in[4];
  const float* bk  = (const float*)d_in[5];
  const float* Wv  = (const float*)d_in[6];
  const float* bv  = (const float*)d_in[7];
  const float* Wo  = (const float*)d_in[8];
  const float* bo  = (const float*)d_in[9];
  const int*   beta = (const int*)d_in[10];
  u16_t* Wb  = (u16_t*)d_ws;     // 4 x 128x128 bf16 = 128 KB
  float* out = (float*)d_out;

  wconv_kernel<<<64, 256, 0, stream>>>(Wq, Wk, Wv, Wo, Wb);
  WindowAttention_21088289423772_kernel<<<4096, 256, 0, stream>>>(
      x, ent, Wb, bq, bk, bv, bo, beta, out);
}
